// BidirectionalMambaBlock_48069273976855
// MI455X (gfx1250) — matrix operand order repaired and checked by reading, not compiled, not run on hardware
//
#include <hip/hip_runtime.h>
#include <hip/hip_bf16.h>
#include <stdint.h>

// ---------------------------------------------------------------------------
// Bidirectional Mamba2 block for gfx1250 (MI455X).
// GEMMs run on v_wmma_f32_16x16x32_bf16 (wave32 WMMA), scan is a register-
// resident sequential recurrence. fp32 in/out, bf16 tensor-core operands.
// ---------------------------------------------------------------------------

typedef __attribute__((ext_vector_type(16))) __bf16        v16bf;
typedef __attribute__((ext_vector_type(8)))  float         v8f;
typedef __attribute__((ext_vector_type(4)))  unsigned int  u32x4;

#define DM     1024   // d_model
#define DI     2048   // d_inner
#define NH     32     // heads
#define HD     64     // head dim
#define NST    64     // d_state
#define CC     2176   // conv channels (DI + 2*NST)
#define DP     4256   // proj dim (2*DI + 2*NST + NH)
#define DPPAD  4288   // DP rounded up to multiple of 64
#define LSEQ   1024
#define TOK    2048   // B_SZ * LSEQ

__device__ __forceinline__ unsigned short f2bf(float f) {
  unsigned int u = __float_as_uint(f);
  unsigned int r = u + 0x7FFFu + ((u >> 16) & 1u);   // round-to-nearest-even
  return (unsigned short)(r >> 16);
}

union Frag16 { u32x4 q[2]; v16bf v; };

// ---------------------------------------------------------------------------
// bf16 WMMA GEMM: C[M,N] = A[M,K] * Bt[N,K]^T (+ bias[n])
// Block = 128 threads = 4 waves; each wave computes a 32x64 C tile:
// 2 A fragments x 4 B fragments -> 8 WMMAs per K-step of 32.
// All 6 fragment loads are issued before the WMMA chain so the loads clause
// together and overlap with the matrix pipe instead of wait-per-wmma.
// A fragment (16-bit 16x32, ISA 7.12.2): lane L, m = L%16,
//   K chunks { (L/16)*8 .. +7 } and { 16 + (L/16)*8 .. +7 }  -> two 16B loads.
// B fragment: lane L, n = L%16, K chunks { (L/16)*8 .. +7 } and
//   { 16 + (L/16)*8 .. +7 } -> two 16B loads from the pre-transposed [N][K] weight.
// ---------------------------------------------------------------------------
__global__ void __launch_bounds__(128)
gemm_bf16_wmma(const unsigned short* __restrict__ A,
               const unsigned short* __restrict__ Bt,
               float* __restrict__ C,
               int M, int N, int K,
               const float* __restrict__ bias) {
  const int wave = threadIdx.x >> 5;
  const int lane = threadIdx.x & 31;
  const int l16  = lane & 15;
  const int hi   = lane >> 4;
  const int m0   = blockIdx.y * 128 + wave * 32;
  const int n0   = blockIdx.x * 64;

  v8f acc[2][4] = {};

  const unsigned short* arow0 = A + (size_t)(m0 + l16) * K;
  const unsigned short* arow1 = A + (size_t)(m0 + 16 + l16) * K;
  const unsigned short* brow  = Bt + (size_t)(n0 + l16) * K + hi * 8;

  for (int k = 0; k < K; k += 32) {
    Frag16 a0, a1, b[4];
    a0.q[0] = *(const u32x4*)(arow0 + k + hi * 8);
    a0.q[1] = *(const u32x4*)(arow0 + k + 16 + hi * 8);
    a1.q[0] = *(const u32x4*)(arow1 + k + hi * 8);
    a1.q[1] = *(const u32x4*)(arow1 + k + 16 + hi * 8);
#pragma unroll
    for (int t = 0; t < 4; ++t) {
      const unsigned short* bp = brow + (size_t)t * 16 * K + k;
      b[t].q[0] = *(const u32x4*)(bp);
      b[t].q[1] = *(const u32x4*)(bp + 16);
    }
    __builtin_prefetch(arow0 + k + 128, 0, 3);   // global_prefetch_b8 (near)
    __builtin_prefetch(arow1 + k + 128, 0, 3);
#pragma unroll
    for (int t = 0; t < 4; ++t) {
      acc[0][t] = __builtin_amdgcn_wmma_f32_16x16x32_bf16(
          false, a0.v, false, b[t].v, (short)0, acc[0][t], false, false);
      acc[1][t] = __builtin_amdgcn_wmma_f32_16x16x32_bf16(
          false, a1.v, false, b[t].v, (short)0, acc[1][t], false, false);
    }
  }

  // C/D layout: lane L -> n = L%16 (+16*t), rows m = (L/16)*8 + r
#pragma unroll
  for (int g = 0; g < 2; ++g) {
#pragma unroll
    for (int t = 0; t < 4; ++t) {
      int n = n0 + t * 16 + l16;
      if (n < N) {
        float bv = bias ? bias[n] : 0.0f;
#pragma unroll
        for (int r = 0; r < 8; ++r) {
          int m = m0 + g * 16 + hi * 8 + r;
          C[(size_t)m * N + n] = acc[g][t][r] + bv;
        }
      }
    }
  }
}

// ---------------------------------------------------------------------------
// x -> bf16 (forward order and sequence-flipped order)
// ---------------------------------------------------------------------------
__global__ void cvt_x_kernel(const float* __restrict__ x,
                             unsigned short* __restrict__ xf,
                             unsigned short* __restrict__ xr) {
  int idx = blockIdx.x * 256 + threadIdx.x;
  if (idx >= TOK * DM) return;
  int row = idx >> 10;            // token
  int d   = idx & 1023;
  int b   = row >> 10;
  int l   = row & 1023;
  xf[idx] = f2bf(x[idx]);
  xr[idx] = f2bf(x[(size_t)((b << 10) + (1023 - l)) * DM + d]);
}

// ---------------------------------------------------------------------------
// W[K,N] fp32 -> Wt[Npad,K] bf16 (zero-filled pad rows)
// ---------------------------------------------------------------------------
__global__ void transpose_bf16_kernel(const float* __restrict__ W,
                                      unsigned short* __restrict__ Wt,
                                      int K, int N, int Npad) {
  long long idx = (long long)blockIdx.x * 256 + threadIdx.x;
  if (idx >= (long long)Npad * K) return;
  int n = (int)(idx / K);
  int k = (int)(idx % K);
  Wt[idx] = (n < N) ? f2bf(W[(size_t)k * N + n]) : (unsigned short)0;
}

// ---------------------------------------------------------------------------
// Causal depthwise conv (width 4) over xBC slice of zxbcdt, + bias, SiLU.
// ---------------------------------------------------------------------------
__global__ void __launch_bounds__(256)
conv_silu_kernel(const float* __restrict__ zx,   // [TOK][DP]
                 const float* __restrict__ cw,   // [4][1][CC]
                 const float* __restrict__ cb,   // [CC]
                 float* __restrict__ out) {      // [TOK][CC]
  const int row = blockIdx.x;
  const int b = row >> 10;
  const int l = row & 1023;
  for (int c = threadIdx.x; c < CC; c += 256) {
    float acc = cb[c];
#pragma unroll
    for (int j = 0; j < 4; ++j) {
      int lj = l - 3 + j;
      if (lj >= 0)
        acc += cw[j * CC + c] * zx[(size_t)((b << 10) + lj) * DP + DI + c];
    }
    out[(size_t)row * CC + c] = acc / (1.0f + __expf(-acc));   // SiLU
  }
}

// ---------------------------------------------------------------------------
// dt = softplus(dt_raw + dtb); dA = exp(dt * (-exp(Alog)))
// ---------------------------------------------------------------------------
__global__ void dt_kernel(const float* __restrict__ zx,
                          const float* __restrict__ dtb,
                          const float* __restrict__ Alog,
                          float* __restrict__ dtv,
                          float* __restrict__ dAv) {
  const int row = blockIdx.x;
  const int h   = threadIdx.x;     // 32 heads
  float d  = zx[(size_t)row * DP + DI + CC + h] + dtb[h];
  float sp = (d > 20.0f) ? d : log1pf(__expf(d));
  float A  = -__expf(Alog[h]);
  dtv[(size_t)row * NH + h] = sp;
  dAv[(size_t)row * NH + h] = __expf(sp * A);
}

// ---------------------------------------------------------------------------
// Sequential selective scan. One block per (batch, head): state h[64x64] =
// 16 regs/thread. Thread t: p = t&63, n-slice = (t>>6)*16 .. +15.
// y_p = sum_n C_n h_{p,n}  (4-way LDS reduction) + D*x_p.
// ---------------------------------------------------------------------------
__global__ void __launch_bounds__(256)
scan_kernel(const float* __restrict__ xbc,    // [TOK][CC] (SiLU'd)
            const float* __restrict__ dtv,    // [TOK][NH]
            const float* __restrict__ dAv,    // [TOK][NH]
            const float* __restrict__ Dsk,    // [NH]
            float* __restrict__ yraw) {       // [TOK][DI]
  const int b  = blockIdx.x >> 5;
  const int hd = blockIdx.x & 31;
  const int t  = threadIdx.x;
  const int p  = t & 63;
  const int nb = (t >> 6) << 4;

  __shared__ float sB[64];
  __shared__ float sC[64];
  __shared__ float sR[256];

  float hreg[16];
#pragma unroll
  for (int i = 0; i < 16; ++i) hreg[i] = 0.0f;
  const float dsk = Dsk[hd];

  for (int l = 0; l < LSEQ; ++l) {
    const int row = (b << 10) + l;
    const size_t base = (size_t)row * CC;
    if (t < 64)        sB[t]      = xbc[base + DI + t];
    else if (t < 128)  sC[t - 64] = xbc[base + DI + NST + (t - 64)];
    __syncthreads();

    const float xp   = xbc[base + hd * HD + p];
    const float da   = dAv[(size_t)row * NH + hd];
    const float coef = dtv[(size_t)row * NH + hd] * xp;
    float acc = 0.0f;
#pragma unroll
    for (int i = 0; i < 16; ++i) {
      hreg[i] = da * hreg[i] + coef * sB[nb + i];
      acc    += sC[nb + i] * hreg[i];
    }
    sR[t] = acc;
    __syncthreads();
    if (t < 64) {
      float y = sR[t] + sR[t + 64] + sR[t + 128] + sR[t + 192] + dsk * xp;
      yraw[(size_t)row * DI + hd * HD + t] = y;
    }
    __syncthreads();
  }
}

// ---------------------------------------------------------------------------
// y *= silu(z); gated RMSNorm over DI; * normw; -> bf16 for the Wout GEMM.
// ---------------------------------------------------------------------------
__global__ void __launch_bounds__(256)
gate_norm_kernel(const float* __restrict__ yraw,
                 const float* __restrict__ zx,
                 const float* __restrict__ normw,
                 unsigned short* __restrict__ ybf) {
  __shared__ float sy[DI];
  __shared__ float sr[256];
  const int row = blockIdx.x;
  const int t = threadIdx.x;
  float sum = 0.0f;
  for (int j = t; j < DI; j += 256) {
    float z = zx[(size_t)row * DP + j];
    float g = z / (1.0f + __expf(-z));
    float v = yraw[(size_t)row * DI + j] * g;
    sy[j] = v;
    sum += v * v;
  }
  sr[t] = sum;
  __syncthreads();
  for (int s = 128; s > 0; s >>= 1) {
    if (t < s) sr[t] += sr[t + s];
    __syncthreads();
  }
  float scale = rsqrtf(sr[0] / (float)DI + 1e-5f);
  for (int j = t; j < DI; j += 256)
    ybf[(size_t)row * DI + j] = f2bf(sy[j] * scale * normw[j]);
}

// ---------------------------------------------------------------------------
// Pack concat([out_fwd, flip(out_rev)]) as bf16 operand for the final GEMM.
// ---------------------------------------------------------------------------
__global__ void pack_kernel(const float* __restrict__ of,
                            const float* __restrict__ ob,
                            unsigned short* __restrict__ outcat) {
  int idx = blockIdx.x * 256 + threadIdx.x;
  if (idx >= TOK * DM) return;
  int row = idx >> 10;
  int j   = idx & 1023;
  int b   = row >> 10;
  int l   = row & 1023;
  outcat[(size_t)row * DI + j] = f2bf(of[idx]);
  int rrow = (b << 10) + (1023 - l);
  outcat[(size_t)row * DI + DM + j] = f2bf(ob[(size_t)rrow * DM + j]);
}

// ---------------------------------------------------------------------------
extern "C" void kernel_launch(void* const* d_in, const int* in_sizes, int n_in,
                              void* d_out, int out_size, void* d_ws, size_t ws_size,
                              hipStream_t stream) {
  const float* x      = (const float*)d_in[0];
  const float* fWin   = (const float*)d_in[1];
  const float* fconvw = (const float*)d_in[2];
  const float* fconvb = (const float*)d_in[3];
  const float* fdtb   = (const float*)d_in[4];
  const float* fAlog  = (const float*)d_in[5];
  const float* fD     = (const float*)d_in[6];
  const float* fnormw = (const float*)d_in[7];
  const float* fWout  = (const float*)d_in[8];
  const float* bWin   = (const float*)d_in[9];
  const float* bconvw = (const float*)d_in[10];
  const float* bconvb = (const float*)d_in[11];
  const float* bdtb   = (const float*)d_in[12];
  const float* bAlog  = (const float*)d_in[13];
  const float* bD     = (const float*)d_in[14];
  const float* bnormw = (const float*)d_in[15];
  const float* bWout  = (const float*)d_in[16];
  const float* Wo     = (const float*)d_in[17];
  const float* bo     = (const float*)d_in[18];
  float* out = (float*)d_out;

  // ---- workspace carve-out (256B aligned) ----
  size_t off = 0;
  char* base = (char*)d_ws;
  auto alloc = [&](size_t bytes) -> char* {
    off = (off + 255) & ~(size_t)255;
    char* p = base + off;
    off += bytes;
    return p;
  };
  unsigned short* xbf_f  = (unsigned short*)alloc((size_t)TOK * DM * 2);
  unsigned short* xbf_r  = (unsigned short*)alloc((size_t)TOK * DM * 2);
  unsigned short* WinT_f = (unsigned short*)alloc((size_t)DPPAD * DM * 2);
  unsigned short* WinT_b = (unsigned short*)alloc((size_t)DPPAD * DM * 2);
  unsigned short* WoutT_f= (unsigned short*)alloc((size_t)DM * DI * 2);
  unsigned short* WoutT_b= (unsigned short*)alloc((size_t)DM * DI * 2);
  unsigned short* WoT    = (unsigned short*)alloc((size_t)DM * DI * 2);
  float* zx_f   = (float*)alloc((size_t)TOK * DP * 4);
  float* zx_b   = (float*)alloc((size_t)TOK * DP * 4);
  float* xbcc_f = (float*)alloc((size_t)TOK * CC * 4);
  float* xbcc_b = (float*)alloc((size_t)TOK * CC * 4);
  float* dtv_f  = (float*)alloc((size_t)TOK * NH * 4);
  float* dtv_b  = (float*)alloc((size_t)TOK * NH * 4);
  float* dAv_f  = (float*)alloc((size_t)TOK * NH * 4);
  float* dAv_b  = (float*)alloc((size_t)TOK * NH * 4);
  float* yraw_f = (float*)alloc((size_t)TOK * DI * 4);
  float* yraw_b = (float*)alloc((size_t)TOK * DI * 4);
  unsigned short* ybf_f  = (unsigned short*)alloc((size_t)TOK * DI * 2);
  unsigned short* ybf_b  = (unsigned short*)alloc((size_t)TOK * DI * 2);
  float* od_f   = (float*)alloc((size_t)TOK * DM * 4);
  float* od_b   = (float*)alloc((size_t)TOK * DM * 4);
  unsigned short* outcat = (unsigned short*)alloc((size_t)TOK * DI * 2);
  (void)ws_size; (void)in_sizes; (void)n_in; (void)out_size;

  // 1) activations + weights to bf16 (weights transposed to [N][K])
  cvt_x_kernel<<<(TOK * DM + 255) / 256, 256, 0, stream>>>(x, xbf_f, xbf_r);
  {
    long long e;
    e = (long long)DPPAD * DM;
    transpose_bf16_kernel<<<(int)((e + 255) / 256), 256, 0, stream>>>(fWin, WinT_f, DM, DP, DPPAD);
    transpose_bf16_kernel<<<(int)((e + 255) / 256), 256, 0, stream>>>(bWin, WinT_b, DM, DP, DPPAD);
    e = (long long)DM * DI;
    transpose_bf16_kernel<<<(int)((e + 255) / 256), 256, 0, stream>>>(fWout, WoutT_f, DI, DM, DM);
    transpose_bf16_kernel<<<(int)((e + 255) / 256), 256, 0, stream>>>(bWout, WoutT_b, DI, DM, DM);
    transpose_bf16_kernel<<<(int)((e + 255) / 256), 256, 0, stream>>>(Wo, WoT, DI, DM, DM);
  }

  // 2) input projections (WMMA): zxbcdt = x @ Win  [TOK x DP]
  {
    dim3 g(DPPAD / 64, TOK / 128);
    gemm_bf16_wmma<<<g, 128, 0, stream>>>(xbf_f, WinT_f, zx_f, TOK, DP, DM, nullptr);
    gemm_bf16_wmma<<<g, 128, 0, stream>>>(xbf_r, WinT_b, zx_b, TOK, DP, DM, nullptr);
  }

  // 3) depthwise causal conv + SiLU; dt/dA precompute
  conv_silu_kernel<<<TOK, 256, 0, stream>>>(zx_f, fconvw, fconvb, xbcc_f);
  conv_silu_kernel<<<TOK, 256, 0, stream>>>(zx_b, bconvw, bconvb, xbcc_b);
  dt_kernel<<<TOK, NH, 0, stream>>>(zx_f, fdtb, fAlog, dtv_f, dAv_f);
  dt_kernel<<<TOK, NH, 0, stream>>>(zx_b, bdtb, bAlog, dtv_b, dAv_b);

  // 4) selective scan (one block per batch x head)
  scan_kernel<<<2 * NH, 256, 0, stream>>>(xbcc_f, dtv_f, dAv_f, fD, yraw_f);
  scan_kernel<<<2 * NH, 256, 0, stream>>>(xbcc_b, dtv_b, dAv_b, bD, yraw_b);

  // 5) gate + RMSNorm -> bf16
  gate_norm_kernel<<<TOK, 256, 0, stream>>>(yraw_f, zx_f, fnormw, ybf_f);
  gate_norm_kernel<<<TOK, 256, 0, stream>>>(yraw_b, zx_b, bnormw, ybf_b);

  // 6) output projections (WMMA)
  {
    dim3 g(DM / 64, TOK / 128);
    gemm_bf16_wmma<<<g, 128, 0, stream>>>(ybf_f, WoutT_f, od_f, TOK, DM, DI, nullptr);
    gemm_bf16_wmma<<<g, 128, 0, stream>>>(ybf_b, WoutT_b, od_b, TOK, DM, DI, nullptr);
  }

  // 7) un-flip backward stream, pack concat operand, final GEMM + bias
  pack_kernel<<<(TOK * DM + 255) / 256, 256, 0, stream>>>(od_f, od_b, outcat);
  {
    dim3 g(DM / 64, TOK / 128);
    gemm_bf16_wmma<<<g, 128, 0, stream>>>(outcat, WoT, out, TOK, DM, DI, bo);
  }
}
